// co_att_q_72808285602149
// MI455X (gfx1250) — matrix operand order repaired and checked by reading, not compiled, not run on hardware
//
#include <hip/hip_runtime.h>
#include <hip/hip_bf16.h>

typedef _Float16 half_t;
typedef __attribute__((ext_vector_type(16))) _Float16 v16h;
typedef __attribute__((ext_vector_type(8)))  _Float16 v8h;
typedef __attribute__((ext_vector_type(8)))  float    v8f;

#define HW     3136
#define WDIM   56
#define PADW   58
#define PADHW  3364   // 58*58
#define CDIM   128
#define LDK    40     // LDS row stride in halfs (32 data + 8 pad)
#define BM     64
#define BN     128
#define BK     32

// D = A(16x32 f16) * B(32x16 f16) + C(16x16 f32), wave32 WMMA
__device__ __forceinline__ v8f wmma_f16(v16h a, v16h b, v8f c) {
  return __builtin_amdgcn_wmma_f32_16x16x32_f16(false, a, false, b, (short)0, c,
                                                false, false);
}

// CDNA5 async copy: 16 bytes global -> LDS, tracked by ASYNCcnt (no VGPR staging).
// LDS byte address = low 32 bits of the generic shared pointer (ISA 10.2).
__device__ __forceinline__ void async_b128(unsigned lds, const void* gptr) {
  asm volatile("global_load_async_to_lds_b128 %0, %1, off"
               :: "v"(lds), "v"(gptr) : "memory");
}
__device__ __forceinline__ void wait_async0() {
  asm volatile("s_wait_asynccnt 0x0" ::: "memory");
}
__device__ __forceinline__ unsigned lds_addr(const half_t* p) {
  return (unsigned)(uintptr_t)p;
}

// A fragment: lane L holds row m = L%16; halfs 0..7 = K kb..kb+7, 8..15 = K 16+kb..
__device__ __forceinline__ v16h frag_a(const half_t* As, int row) {
  int lane = threadIdx.x & 31;
  const half_t* p = As + (row + (lane & 15)) * LDK + ((lane >> 4) << 3);
  v8h lo = *(const v8h*)p;
  v8h hi = *(const v8h*)(p + 16);
  v16h r;
#pragma unroll
  for (int i = 0; i < 8; ++i) { r[i] = lo[i]; r[i + 8] = hi[i]; }
  return r;
}

// B fragment: lane L holds column n = L%16; halfs 0..7 = K kb..kb+7, 8..15 = K 16+kb.. (kb = 8*(L/16))
__device__ __forceinline__ v16h frag_b(const half_t* Bs, int col) {
  int lane = threadIdx.x & 31;
  const half_t* p = Bs + (col + (lane & 15)) * LDK + ((lane >> 4) << 3);
  v8h lo = *(const v8h*)p;
  v8h hi = *(const v8h*)(p + 16);
  v16h r;
#pragma unroll
  for (int i = 0; i < 8; ++i) { r[i] = lo[i]; r[i + 8] = hi[i]; }
  return r;
}

// -------- BN fold --------
__global__ void fold_bn_kernel(const float* __restrict__ g, const float* __restrict__ beta,
                               const float* __restrict__ mean, const float* __restrict__ var,
                               const float* __restrict__ bconv,
                               float* __restrict__ scale, float* __restrict__ bias, int C) {
  int i = threadIdx.x + blockIdx.x * blockDim.x;
  if (i < C) {
    float s = g[i] * rsqrtf(var[i] + 1e-5f);
    scale[i] = s;
    float bc = bconv ? bconv[i] : 0.f;
    bias[i] = bc * s + beta[i] - mean[i] * s;
  }
}

// weights: OIHW -> [oc][(r,s)][ic] f16, folded BN scale. KK = KH*KW (9 or 1).
__global__ void cvt_weights_kernel(const float* __restrict__ w, const float* __restrict__ scale,
                                   half_t* __restrict__ out, int IC, int KK, int Ktot,
                                   int total) {
  int i = threadIdx.x + blockIdx.x * blockDim.x;
  if (i < total) {
    int oc = i / Ktot;
    int kr = i - oc * Ktot;
    int rs = kr / IC;
    int ic = kr - rs * IC;
    float s = scale ? scale[oc] : 1.f;
    out[i] = (half_t)(w[(long long)oc * Ktot + ic * KK + rs] * s);
  }
}

// NCHW f32 -> padded channel-last [B][58][58][C] f16 (halo pre-zeroed by memset).
__global__ __launch_bounds__(256) void nchw_to_clpad_kernel(const float* __restrict__ in,
                                                            half_t* __restrict__ out, int C) {
  __shared__ float tile[32][33];
  int b = blockIdx.z;
  int p0 = blockIdx.x * 32;
  int c0 = blockIdx.y * 32;
  int tx = threadIdx.x, ty = threadIdx.y;
#pragma unroll
  for (int i = 0; i < 32; i += 8)
    tile[ty + i][tx] = in[((long long)b * C + c0 + ty + i) * HW + p0 + tx];
  __syncthreads();
#pragma unroll
  for (int i = 0; i < 32; i += 8) {
    int p = p0 + ty + i;
    int yy = p / WDIM;
    int xx = p - yy * WDIM;
    out[((long long)b * PADHW + (yy + 1) * PADW + (xx + 1)) * C + c0 + tx] =
        (half_t)tile[tx][ty + i];
  }
}

// -------- implicit-GEMM conv, OC=128, padded channel-last input --------
// Async double-buffered pipeline: tile k+1 streams into LDS while tile k is computed.
// outMode: 0 = channel-major [128][Ncols], 1 = channel-last [Ncols][128],
//          2 = padded channel-last [B][58][58][128]
template <int KH>
__global__ __launch_bounds__(256) void conv_gemm_kernel(
    const half_t* __restrict__ in, const half_t* __restrict__ wts,
    const float* __restrict__ bias, half_t* __restrict__ out,
    int icShift, int Ncols, int relu, int outMode) {
  const int IC = 1 << icShift;
  const int TAPS = KH * KH;
  const int Ktot = TAPS * IC;
  __shared__ half_t As[2 * BM * LDK];
  __shared__ half_t Bs[2 * BN * LDK];
  int t = threadIdx.x;
  int m0 = blockIdx.y * BM;
  int n0 = blockIdx.x * BN;
  int w = t >> 5;
  int wm = (w >> 2) * 32;
  int wn = (w & 3) * 32;
  v8f acc00 = {}, acc01 = {}, acc10 = {}, acc11 = {};

  // A loader: 8 contiguous halfs per thread, pointer marches linearly in k
  int aRow = (t * 8) >> 5;
  int aK = (t * 8) & 31;
  const half_t* aPtr = wts + (long long)(m0 + aRow) * Ktot + aK;

  // B loader: 2 threads per pixel column, 16 contiguous ic each; tap bases hoisted
  int j = t >> 1;
  int ks = (t & 1) * 16;
  int gn_l = n0 + j;
  int bb = gn_l / HW;
  int rem = gn_l - bb * HW;
  int y = rem / WDIM;
  int x = rem - y * WDIM;

  const half_t* bPtr[TAPS + 1];
#pragma unroll
  for (int rs = 0; rs < TAPS; ++rs) {
    int r = (KH == 3) ? (rs / 3) : 1;
    int s = (KH == 3) ? (rs % 3) : 1;
    bPtr[rs] = in + ((long long)bb * PADHW + (y + r) * PADW + (x + s)) * IC + ks;
  }
  bPtr[TAPS] = bPtr[TAPS - 1];

  unsigned aL0 = lds_addr(As + aRow * LDK + aK);
  unsigned aL1 = aL0 + BM * LDK * sizeof(half_t);
  unsigned bL0 = lds_addr(Bs + j * LDK + ks);
  unsigned bL1 = bL0 + BN * LDK * sizeof(half_t);

  // prologue: stream first tile into buffer 0
  async_b128(aL0, aPtr);
  async_b128(bL0, bPtr[0]);
  async_b128(bL0 + 16, bPtr[0] + 8);

  int par = 0;
#pragma unroll
  for (int rs = 0; rs < TAPS; ++rs) {
    for (int ic0 = 0; ic0 < IC; ic0 += BK) {
      wait_async0();
      __syncthreads();
      bool last = (rs == TAPS - 1) && (ic0 + BK >= IC);
      if (!last) {  // stream next tile into the other buffer, overlapping WMMA
        aPtr += BK;
        const half_t* nb = (ic0 + BK < IC) ? (bPtr[rs] + ic0 + BK) : bPtr[rs + 1];
        unsigned na = par ? aL0 : aL1;
        unsigned nbl = par ? bL0 : bL1;
        async_b128(na, aPtr);
        async_b128(nbl, nb);
        async_b128(nbl + 16, nb + 8);
      }
      const half_t* Ab = As + (par ? BM * LDK : 0);
      const half_t* Bb = Bs + (par ? BN * LDK : 0);
      v16h a0 = frag_a(Ab, wm);
      v16h a1 = frag_a(Ab, wm + 16);
      v16h b0 = frag_b(Bb, wn);
      v16h b1 = frag_b(Bb, wn + 16);
      acc00 = wmma_f16(a0, b0, acc00);
      acc01 = wmma_f16(a0, b1, acc01);
      acc10 = wmma_f16(a1, b0, acc10);
      acc11 = wmma_f16(a1, b1, acc11);
      par ^= 1;
    }
  }

  int lane = t & 31;
  int cn = lane & 15;
  int rm = (lane >> 4) << 3;
  v8f accs[4] = {acc00, acc01, acc10, acc11};
#pragma unroll
  for (int sm = 0; sm < 2; ++sm)
#pragma unroll
    for (int sn = 0; sn < 2; ++sn) {
      v8f acc = accs[sm * 2 + sn];
      int gn = n0 + wn + sn * 16 + cn;
      int obb = gn / HW;
      int orem = gn - obb * HW;
      int oy = orem / WDIM;
      int ox = orem - oy * WDIM;
      long long clIdx = (long long)gn * CDIM;
      long long padIdx = ((long long)obb * PADHW + (oy + 1) * PADW + (ox + 1)) * CDIM;
#pragma unroll
      for (int r = 0; r < 8; ++r) {
        int gm = m0 + wm + sm * 16 + rm + r;
        float v = acc[r];
        if (bias) v += bias[gm];
        if (relu) v = fmaxf(v, 0.f);
        half_t h = (half_t)v;
        if (outMode == 0)      out[(long long)gm * Ncols + gn] = h;
        else if (outMode == 1) out[clIdx + gm] = h;
        else                   out[padIdx + gm] = h;
      }
    }
}

// -------- attention scores: A[b][n1][n2] = sum_c q1[n1][c] * q2[n2][c], f16 out --------
__global__ __launch_bounds__(256) void attn_gemm_kernel(
    const half_t* __restrict__ q1nc, const half_t* __restrict__ q2nc,
    half_t* __restrict__ Aout, int Npix) {
  __shared__ half_t As[2 * BM * LDK];
  __shared__ half_t Bs[2 * BN * LDK];
  int t = threadIdx.x;
  int bz = blockIdx.z;
  int m0 = blockIdx.y * BM;
  int n0 = blockIdx.x * BN;
  int w = t >> 5;
  int wm = (w >> 2) * 32;
  int wn = (w & 3) * 32;
  v8f acc00 = {}, acc01 = {}, acc10 = {}, acc11 = {};

  int aRow = (t * 8) >> 5;
  int aK = (t * 8) & 31;
  const half_t* aPtr = q1nc + (long long)(bz * Npix + m0 + aRow) * CDIM + aK;

  int j = t >> 1;
  int ks = (t & 1) * 16;
  int gn_l = n0 + j;
  // clamp over-tile columns to col 0: harmless data, discarded by the store guard
  const half_t* bPtr = q2nc + ((long long)bz * Npix + (gn_l < Npix ? gn_l : 0)) * CDIM + ks;

  unsigned aL0 = lds_addr(As + aRow * LDK + aK);
  unsigned aL1 = aL0 + BM * LDK * sizeof(half_t);
  unsigned bL0 = lds_addr(Bs + j * LDK + ks);
  unsigned bL1 = bL0 + BN * LDK * sizeof(half_t);

  async_b128(aL0, aPtr);
  async_b128(bL0, bPtr);
  async_b128(bL0 + 16, bPtr + 8);

  int par = 0;
  for (int k0 = 0; k0 < CDIM; k0 += BK) {
    wait_async0();
    __syncthreads();
    if (k0 + BK < CDIM) {
      aPtr += BK;
      bPtr += BK;
      unsigned na = par ? aL0 : aL1;
      unsigned nbl = par ? bL0 : bL1;
      async_b128(na, aPtr);
      async_b128(nbl, bPtr);
      async_b128(nbl + 16, bPtr + 8);
    }
    const half_t* Ab = As + (par ? BM * LDK : 0);
    const half_t* Bb = Bs + (par ? BN * LDK : 0);
    v16h a0 = frag_a(Ab, wm);
    v16h a1 = frag_a(Ab, wm + 16);
    v16h b0 = frag_b(Bb, wn);
    v16h b1 = frag_b(Bb, wn + 16);
    acc00 = wmma_f16(a0, b0, acc00);
    acc01 = wmma_f16(a0, b1, acc01);
    acc10 = wmma_f16(a1, b0, acc10);
    acc11 = wmma_f16(a1, b1, acc11);
    par ^= 1;
  }

  int lane = t & 31;
  int cn = lane & 15;
  int rm = (lane >> 4) << 3;
  v8f accs[4] = {acc00, acc01, acc10, acc11};
  long long base = (long long)bz * Npix * Npix;
#pragma unroll
  for (int sm = 0; sm < 2; ++sm)
#pragma unroll
    for (int sn = 0; sn < 2; ++sn) {
      v8f acc = accs[sm * 2 + sn];
      int gn = n0 + wn + sn * 16 + cn;
      if (gn >= Npix) continue;
#pragma unroll
      for (int r = 0; r < 8; ++r) {
        int gm = m0 + wm + sm * 16 + rm + r;
        Aout[base + (long long)gm * Npix + gn] = (half_t)acc[r];
      }
    }
}

// softmax over n1 (axis=1): one thread per column m, coalesced across m
__global__ __launch_bounds__(256) void softmax_col_kernel(half_t* __restrict__ A, int Npix) {
  int m = blockIdx.x * blockDim.x + threadIdx.x;
  int b = blockIdx.y;
  if (m >= Npix) return;
  half_t* base = A + (long long)b * Npix * Npix + m;
  float mx = -1e30f, sm = 0.f;
  for (int n = 0; n < Npix; ++n) {
    float v = (float)base[(long long)n * Npix];
    float nm = fmaxf(mx, v);
    sm = sm * __expf(mx - nm) + __expf(v - nm);
    mx = nm;
  }
  float inv = 1.f / sm;
  for (int n = 0; n < Npix; ++n) {
    long long idx = (long long)n * Npix;
    float v = (float)base[idx];
    base[idx] = (half_t)(__expf(v - mx) * inv);
  }
}

// softmax over n2 (axis=2): one block per row, in place
__global__ __launch_bounds__(256) void softmax_row_kernel(half_t* __restrict__ A, int Npix) {
  int n = blockIdx.x;
  int b = blockIdx.y;
  half_t* row = A + ((long long)b * Npix + n) * Npix;
  __shared__ float red[256];
  float mx = -1e30f;
  for (int m = threadIdx.x; m < Npix; m += 256) mx = fmaxf(mx, (float)row[m]);
  red[threadIdx.x] = mx;
  __syncthreads();
  for (int s = 128; s > 0; s >>= 1) {
    if (threadIdx.x < s) red[threadIdx.x] = fmaxf(red[threadIdx.x], red[threadIdx.x + s]);
    __syncthreads();
  }
  mx = red[0];
  __syncthreads();
  float sm = 0.f;
  for (int m = threadIdx.x; m < Npix; m += 256) sm += __expf((float)row[m] - mx);
  red[threadIdx.x] = sm;
  __syncthreads();
  for (int s = 128; s > 0; s >>= 1) {
    if (threadIdx.x < s) red[threadIdx.x] += red[threadIdx.x + s];
    __syncthreads();
  }
  sm = red[0];
  __syncthreads();
  float inv = 1.f / sm;
  for (int m = threadIdx.x; m < Npix; m += 256)
    row[m] = (half_t)(__expf((float)row[m] - mx) * inv);
}

// -------- final: out[b][c][n1] = sum_m x2v[c][b*N+m] * Anorm[b][n1][m] + x1 --------
__global__ __launch_bounds__(256) void final_gemm_kernel(
    const half_t* __restrict__ x2v, const half_t* __restrict__ An,
    const float* __restrict__ x1, float* __restrict__ out, int Npix, int BNtot) {
  __shared__ half_t As[2 * BM * LDK];
  __shared__ half_t Bs[2 * BN * LDK];
  int t = threadIdx.x;
  int bz = blockIdx.z;
  int m0 = blockIdx.y * BM;  // over C
  int n0 = blockIdx.x * BN;  // over n1
  int w = t >> 5;
  int wm = (w >> 2) * 32;
  int wn = (w & 3) * 32;
  v8f acc00 = {}, acc01 = {}, acc10 = {}, acc11 = {};

  int aRow = (t * 8) >> 5;
  int aK = (t * 8) & 31;
  const half_t* aPtr = x2v + (long long)(m0 + aRow) * BNtot + bz * Npix + aK;

  int j = t >> 1;
  int ks = (t & 1) * 16;
  int gn1 = n0 + j;
  const half_t* bPtr = An + ((long long)bz * Npix + (gn1 < Npix ? gn1 : 0)) * Npix + ks;

  unsigned aL0 = lds_addr(As + aRow * LDK + aK);
  unsigned aL1 = aL0 + BM * LDK * sizeof(half_t);
  unsigned bL0 = lds_addr(Bs + j * LDK + ks);
  unsigned bL1 = bL0 + BN * LDK * sizeof(half_t);

  async_b128(aL0, aPtr);
  async_b128(bL0, bPtr);
  async_b128(bL0 + 16, bPtr + 8);

  int par = 0;
  for (int k0 = 0; k0 < Npix; k0 += BK) {
    wait_async0();
    __syncthreads();
    if (k0 + BK < Npix) {
      aPtr += BK;
      bPtr += BK;
      unsigned na = par ? aL0 : aL1;
      unsigned nbl = par ? bL0 : bL1;
      async_b128(na, aPtr);
      async_b128(nbl, bPtr);
      async_b128(nbl + 16, bPtr + 8);
    }
    const half_t* Ab = As + (par ? BM * LDK : 0);
    const half_t* Bb = Bs + (par ? BN * LDK : 0);
    v16h a0 = frag_a(Ab, wm);
    v16h a1 = frag_a(Ab, wm + 16);
    v16h b0 = frag_b(Bb, wn);
    v16h b1 = frag_b(Bb, wn + 16);
    acc00 = wmma_f16(a0, b0, acc00);
    acc01 = wmma_f16(a0, b1, acc01);
    acc10 = wmma_f16(a1, b0, acc10);
    acc11 = wmma_f16(a1, b1, acc11);
    par ^= 1;
  }

  int lane = t & 31;
  int cn = lane & 15;
  int rm = (lane >> 4) << 3;
  v8f accs[4] = {acc00, acc01, acc10, acc11};
#pragma unroll
  for (int sm = 0; sm < 2; ++sm)
#pragma unroll
    for (int sn = 0; sn < 2; ++sn) {
      v8f acc = accs[sm * 2 + sn];
      int gn = n0 + wn + sn * 16 + cn;
      if (gn >= Npix) continue;
#pragma unroll
      for (int r = 0; r < 8; ++r) {
        int gm = m0 + wm + sm * 16 + rm + r;
        long long o = ((long long)bz * CDIM + gm) * HW + gn;
        out[o] = acc[r] + x1[o];
      }
    }
}

extern "C" void kernel_launch(void* const* d_in, const int* in_sizes, int n_in,
                              void* d_out, int out_size, void* d_ws, size_t ws_size,
                              hipStream_t stream) {
  (void)in_sizes; (void)n_in; (void)out_size; (void)ws_size;
  const float* x1     = (const float*)d_in[0];
  const float* x2     = (const float*)d_in[1];
  const float* ws_w   = (const float*)d_in[2];
  const float* bs     = (const float*)d_in[3];
  const float* sg     = (const float*)d_in[4];
  const float* sbeta  = (const float*)d_in[5];
  const float* smean  = (const float*)d_in[6];
  const float* svar   = (const float*)d_in[7];
  const float* wq1    = (const float*)d_in[8];
  const float* wq2    = (const float*)d_in[9];
  // d_in[10..15] = v1* : dead code in the reference (x1_v never used) — skipped.
  const float* v2w    = (const float*)d_in[16];
  const float* v2b    = (const float*)d_in[17];
  const float* v2g    = (const float*)d_in[18];
  const float* v2beta = (const float*)d_in[19];
  const float* v2mean = (const float*)d_in[20];
  const float* v2var  = (const float*)d_in[21];
  float* out = (float*)d_out;

  const int B = 4, C = CDIM, N = HW, BNtot = B * N;  // 12544

  size_t off = 0;
  auto carve = [&](size_t bytes) -> char* {
    char* p = (char*)d_ws + off;
    off += (bytes + 255) & ~(size_t)255;
    return p;
  };
  half_t* ws_h    = (half_t*)carve((size_t)C * 4608 * 2);
  half_t* v2w0_h  = (half_t*)carve((size_t)C * 1152 * 2);
  half_t* v2w1_h  = (half_t*)carve((size_t)C * 1152 * 2);
  half_t* wq1_h   = (half_t*)carve((size_t)C * 128 * 2);
  half_t* wq2_h   = (half_t*)carve((size_t)C * 128 * 2);
  float*  scale_s = (float*)carve(C * 4);
  float*  bias_s  = (float*)carve(C * 4);
  float*  scale_v0= (float*)carve(C * 4);
  float*  bias_v0 = (float*)carve(C * 4);
  float*  scale_v1= (float*)carve(C * 4);
  float*  bias_v1 = (float*)carve(C * 4);
  size_t padC   = (size_t)B * PADHW * C * 2;    // 3.44 MB
  size_t pad512 = (size_t)B * PADHW * 512 * 2;  // 13.8 MB
  half_t* x1tp    = (half_t*)carve(padC);
  half_t* x2tp    = (half_t*)carve(pad512);
  half_t* x2sp    = (half_t*)carve(padC);
  half_t* vtmpp   = (half_t*)carve(padC);
  half_t* q1nc    = (half_t*)carve((size_t)BNtot * C * 2);
  half_t* q2nc    = (half_t*)carve((size_t)BNtot * C * 2);
  half_t* x2v     = (half_t*)carve((size_t)C * BNtot * 2);
  half_t* Amat    = (half_t*)carve((size_t)B * N * N * 2);  // 78.7 MB, L2-resident

  // zero halos of padded buffers (interiors overwritten below)
  hipMemsetAsync(x1tp, 0, padC, stream);
  hipMemsetAsync(x2tp, 0, pad512, stream);
  hipMemsetAsync(x2sp, 0, padC, stream);
  hipMemsetAsync(vtmpp, 0, padC, stream);

  // BN folding + weight conversion (k reordered to rs*IC + ic)
  fold_bn_kernel<<<1, 128, 0, stream>>>(sg, sbeta, smean, svar, bs, scale_s, bias_s, C);
  fold_bn_kernel<<<1, 128, 0, stream>>>(v2g, v2beta, v2mean, v2var, v2b,
                                        scale_v0, bias_v0, C);
  fold_bn_kernel<<<1, 128, 0, stream>>>(v2g + C, v2beta + C, v2mean + C, v2var + C,
                                        v2b + C, scale_v1, bias_v1, C);
  int tw = C * 4608;
  cvt_weights_kernel<<<(tw + 255) / 256, 256, 0, stream>>>(ws_w, scale_s, ws_h, 512, 9, 4608, tw);
  tw = C * 1152;
  cvt_weights_kernel<<<(tw + 255) / 256, 256, 0, stream>>>(v2w, scale_v0, v2w0_h, 128, 9, 1152, tw);
  cvt_weights_kernel<<<(tw + 255) / 256, 256, 0, stream>>>(v2w + tw, scale_v1, v2w1_h, 128, 9, 1152, tw);
  tw = C * 128;
  cvt_weights_kernel<<<(tw + 255) / 256, 256, 0, stream>>>(wq1, nullptr, wq1_h, 128, 1, 128, tw);
  cvt_weights_kernel<<<(tw + 255) / 256, 256, 0, stream>>>(wq2, nullptr, wq2_h, 128, 1, 128, tw);

  // activations -> padded channel-last f16
  nchw_to_clpad_kernel<<<dim3(HW / 32, C / 32, B), dim3(32, 8), 0, stream>>>(x1, x1tp, C);
  nchw_to_clpad_kernel<<<dim3(HW / 32, 512 / 32, B), dim3(32, 8), 0, stream>>>(x2, x2tp, 512);

  dim3 blk(256);
  dim3 gconv(BNtot / BN, C / BM, 1);  // (98, 2), exact

  conv_gemm_kernel<3><<<gconv, blk, 0, stream>>>(x2tp, ws_h, bias_s, x2sp, 9, BNtot, 1, 2);
  conv_gemm_kernel<1><<<gconv, blk, 0, stream>>>(x2sp, wq2_h, nullptr, q2nc, 7, BNtot, 0, 1);
  conv_gemm_kernel<1><<<gconv, blk, 0, stream>>>(x1tp, wq1_h, nullptr, q1nc, 7, BNtot, 0, 1);
  conv_gemm_kernel<3><<<gconv, blk, 0, stream>>>(x2sp, v2w0_h, bias_v0, vtmpp, 7, BNtot, 1, 2);
  conv_gemm_kernel<3><<<gconv, blk, 0, stream>>>(vtmpp, v2w1_h, bias_v1, x2v, 7, BNtot, 1, 0);

  dim3 gattn((N + BN - 1) / BN, N / BM, B);  // (25, 49, 4)
  attn_gemm_kernel<<<gattn, blk, 0, stream>>>(q1nc, q2nc, Amat, N);

  softmax_col_kernel<<<dim3((N + 255) / 256, B), blk, 0, stream>>>(Amat, N);
  softmax_row_kernel<<<dim3(N, B), blk, 0, stream>>>(Amat, N);

  dim3 gfin((N + BN - 1) / BN, C / BM, B);  // (25, 2, 4)
  final_gemm_kernel<<<gfin, blk, 0, stream>>>(x2v, Amat, x1, out, N, BNtot);

  hipMemcpyAsync(out + (size_t)B * C * HW, x2,
                 (size_t)B * 512 * HW * sizeof(float),
                 hipMemcpyDeviceToDevice, stream);
}
